// Qwen3VLAttention_13932873908523
// MI455X (gfx1250) — matrix operand order repaired and checked by reading, not compiled, not run on hardware
//
#include <hip/hip_runtime.h>

// ---------------------------------------------------------------------------
// Qwen3-VL attention layer for MI455X (gfx1250, wave32, WMMA + TDM).
// fp32 in/out; bf16 WMMA compute; Tensor Data Mover stages all GEMM tiles.
// ---------------------------------------------------------------------------

typedef __attribute__((ext_vector_type(16))) __bf16       v16bf;
typedef __attribute__((ext_vector_type(8)))  float        v8f;
typedef __attribute__((ext_vector_type(4)))  unsigned int u32x4;
typedef __attribute__((ext_vector_type(8)))  int          i32x8;
typedef __attribute__((ext_vector_type(4)))  int          i32x4;

#define S_LEN   4096
#define HID     2048
#define N_HEADS 16
#define N_KV    4
#define HDIM    128

// ===========================================================================
// TDM: issue a 2D bf16 tile load global -> LDS (Tensor DMA Descriptor D#).
// Rows of `tile_d0` elements, `tile_d1` rows, row stride `stride_el` elements.
// Optional LDS pad skew: insert (pad_amt+1) DWORDs every 2^(pad_iv+1) DWORDs.
// ===========================================================================
__device__ __forceinline__ void tdm_load_tile_bf16(
    unsigned lds_off, const __bf16* gptr,
    unsigned tensor_d0, unsigned tensor_d1,
    unsigned tile_d0, unsigned tile_d1,
    unsigned long long stride_el,
    int pad_iv, int pad_amt)
{
  const unsigned long long ga = (unsigned long long)(size_t)gptr;
  u32x4 g0;
  g0[0] = 1u;                                            // count=1, flags=0
  g0[1] = lds_off;                                       // lds_addr [63:32]
  g0[2] = (unsigned)(ga & 0xFFFFFFFFull);                // global_addr lo
  g0[3] = (unsigned)((ga >> 32) & 0x1FFFFFFull) | (2u << 30);  // hi | type=2

  i32x8 g1;
  unsigned w0 = (1u << 16);                              // data_size=1 (2B)
  if (pad_amt >= 0)
    w0 |= (1u << 20) | ((unsigned)pad_iv << 22) | ((unsigned)pad_amt << 25);
  g1[0] = (int)w0;
  g1[1] = (int)((tensor_d0 & 0xFFFFu) << 16);            // tensor_dim0 lo16
  g1[2] = (int)(((tensor_d0 >> 16) & 0xFFFFu) | ((tensor_d1 & 0xFFFFu) << 16));
  g1[3] = (int)(((tensor_d1 >> 16) & 0xFFFFu) | ((tile_d0 & 0xFFFFu) << 16));
  g1[4] = (int)(tile_d1 & 0xFFFFu);                      // tile_dim1, tile_dim2=0
  g1[5] = (int)(unsigned)(stride_el & 0xFFFFFFFFull);    // tensor_dim0_stride
  g1[6] = (int)(unsigned)((stride_el >> 32) & 0xFFFFull);
  g1[7] = 0;

  const i32x4 z4 = {0, 0, 0, 0};
#if __clang_major__ >= 23
  const i32x8 z8 = {0, 0, 0, 0, 0, 0, 0, 0};
  __builtin_amdgcn_tensor_load_to_lds(g0, g1, z4, z4, z8, 0);
#else
  __builtin_amdgcn_tensor_load_to_lds(g0, g1, z4, z4, 0);
#endif
}

__device__ __forceinline__ unsigned lds_off_of(const void* p) {
  return (unsigned)(size_t)p;   // LDS aperture: low 32 bits = byte offset
}

// ===========================================================================
// fp32 -> bf16 elementwise convert.
// ===========================================================================
__global__ __launch_bounds__(256) void cvt_bf16(
    const float* __restrict__ in, __bf16* __restrict__ out, int n)
{
  const int i = (blockIdx.x * 256 + threadIdx.x) * 4;
  if (i >= n) return;
#pragma unroll
  for (int j = 0; j < 4; ++j) out[i + j] = (__bf16)in[i + j];
}

// ===========================================================================
// fp32 [K,N] -> bf16 transposed [N,K] (32x32 LDS tile, coalesced both sides).
// ===========================================================================
__global__ __launch_bounds__(256) void transpose_cvt_bf16(
    const float* __restrict__ W, __bf16* __restrict__ Wt, int Kd, int Nd)
{
  __shared__ float tile[32][33];
  const int n0 = blockIdx.x * 32, k0 = blockIdx.y * 32;
  const int tx = threadIdx.x & 31, ty = threadIdx.x >> 5;
#pragma unroll
  for (int i = ty; i < 32; i += 8)
    tile[i][tx] = W[(size_t)(k0 + i) * Nd + n0 + tx];
  __syncthreads();
#pragma unroll
  for (int i = ty; i < 32; i += 8)
    Wt[(size_t)(n0 + i) * Kd + k0 + tx] = (__bf16)tile[tx][i];
}

// ===========================================================================
// GEMM: C[M,N] = A[M,K] * Bt[N,K]^T  (bf16 in, CT out).
// 128x128 block tile, K-step 32, double-buffered TDM staging.
// 8 waves in 4x2; each wave owns a 32x64 tile (2x4 WMMA accums) ->
// 8 WMMAs per 12 ds_load_b128 fragment fetches.
// LDS rows: 32 bf16 (64B) padded by TDM to 72B -> pitch 36 elems (bank skew).
// ===========================================================================
#define TM 128
#define TN 128
#define TK 32
#define GP 36   // padded LDS row pitch in bf16 elements

template <typename CT>
__global__ __launch_bounds__(256) void gemm_bf16_tdm(
    const __bf16* __restrict__ A, const __bf16* __restrict__ Bt,
    CT* __restrict__ C, int M, int N, int K)
{
  __shared__ __bf16 lA[2][TM * GP];
  __shared__ __bf16 lB[2][TN * GP];

  const int tid  = threadIdx.x;
  const int wave = tid >> 5;
  const int lane = tid & 31;
  const int lh   = lane & 15;
  const int hi   = lane >> 4;
  const int wm   = (wave >> 1) * 32;   // 4 waves down M
  const int wn   = (wave & 1) * 64;    // 2 waves across N
  const int m0   = blockIdx.y * TM;
  const int n0   = blockIdx.x * TN;

  v8f acc[2][4];
#pragma unroll
  for (int i = 0; i < 2; ++i)
#pragma unroll
    for (int j = 0; j < 4; ++j)
#pragma unroll
      for (int r = 0; r < 8; ++r) acc[i][j][r] = 0.f;

  const int nk = K / TK;
  // prologue: tile 0 into buffer 0  (pad: every 16 DW insert 2 DW -> iv=3, amt=1)
  if (wave == 0) {
    tdm_load_tile_bf16(lds_off_of(lA[0]), A + (size_t)m0 * K,
                       K, M, TK, TM, (unsigned long long)K, 3, 1);
    tdm_load_tile_bf16(lds_off_of(lB[0]), Bt + (size_t)n0 * K,
                       K, N, TK, TN, (unsigned long long)K, 3, 1);
  }

  for (int ki = 0; ki < nk; ++ki) {
    const int buf = ki & 1;
    if (wave == 0) {
      if (ki + 1 < nk) {
        const int k1 = (ki + 1) * TK;
        tdm_load_tile_bf16(lds_off_of(lA[buf ^ 1]), A + (size_t)m0 * K + k1,
                           K, M, TK, TM, (unsigned long long)K, 3, 1);
        tdm_load_tile_bf16(lds_off_of(lB[buf ^ 1]), Bt + (size_t)n0 * K + k1,
                           K, N, TK, TN, (unsigned long long)K, 3, 1);
        __builtin_amdgcn_s_wait_tensorcnt(2);
      } else {
        __builtin_amdgcn_s_wait_tensorcnt(0);
      }
    }
    __syncthreads();   // tile `buf` resident for all waves

    v16bf af[2], bfr[4];
#pragma unroll
    for (int ti = 0; ti < 2; ++ti) {
      const __bf16* p = &lA[buf][(wm + ti * 16 + lh) * GP + hi * 16];
#pragma unroll
      for (int j = 0; j < 8; ++j) { af[ti][j] = p[j]; af[ti][8 + j] = p[8 + j]; }
    }
#pragma unroll
    for (int tj = 0; tj < 4; ++tj) {
      const __bf16* p = &lB[buf][(wn + tj * 16 + lh) * GP + hi * 16];
#pragma unroll
      for (int j = 0; j < 16; ++j) bfr[tj][j] = p[j];
    }
#pragma unroll
    for (int ti = 0; ti < 2; ++ti)
#pragma unroll
      for (int tj = 0; tj < 4; ++tj)
        acc[ti][tj] = __builtin_amdgcn_wmma_f32_16x16x32_bf16(
            false, af[ti], false, bfr[tj], (short)0, acc[ti][tj], false, false);
    __syncthreads();   // all waves done with `buf` before TDM refills it
  }

#pragma unroll
  for (int ti = 0; ti < 2; ++ti)
#pragma unroll
    for (int tj = 0; tj < 4; ++tj)
#pragma unroll
      for (int r = 0; r < 8; ++r) {
        const int row = m0 + wm + ti * 16 + hi * 8 + r;
        const int col = n0 + wn + tj * 16 + lh;
        C[(size_t)row * N + col] = (CT)acc[ti][tj][r];
      }
}

// ===========================================================================
// RoPE in place on bf16 [S, nheads*128].
// ===========================================================================
__global__ void rope_kernel(__bf16* __restrict__ x, const int* __restrict__ pos,
                            int nheads)
{
  const int idx = blockIdx.x * blockDim.x + threadIdx.x;
  if (idx >= S_LEN * nheads * (HDIM / 2)) return;
  const int d = idx & 63;
  const int h = (idx >> 6) % nheads;
  const int s = idx / (64 * nheads);
  const float p   = (float)pos[s];
  const float inv = __powf(1.0e6f, -(float)(2 * d) / (float)HDIM);
  float sn, cs;
  __sincosf(p * inv, &sn, &cs);
  __bf16* row = x + (size_t)s * nheads * HDIM + h * HDIM;
  const float x1 = (float)row[d], x2 = (float)row[d + 64];
  row[d]      = (__bf16)(x1 * cs - x2 * sn);
  row[d + 64] = (__bf16)(x2 * cs + x1 * sn);
}

// ===========================================================================
// Flash attention (causal, GQA rep=4), bf16 in, bf16 out.
// Block = 1 head x 128 query rows (8 waves x 16). K tiles staged by TDM
// (256B rows padded to 272B -> pitch 136), V staged transposed by threads.
// ===========================================================================
#define KP 136   // K-tile LDS pitch (elems)
#define VP 36    // V/P LDS pitch (elems)

__global__ __launch_bounds__(256) void flash_attn_wmma(
    const __bf16* __restrict__ Q, const __bf16* __restrict__ Kv,
    const __bf16* __restrict__ Vv, __bf16* __restrict__ O)
{
  __shared__ __bf16 lK[32 * KP];        // (key, d)  TDM-staged, pad-skewed
  __shared__ __bf16 lVt[HDIM * VP];     // (d, key)  thread-staged transpose
  __shared__ __bf16 lP[8][16 * VP];     // per-wave P (C-layout -> A-layout)

  const int tid  = threadIdx.x;
  const int wave = tid >> 5;
  const int lane = tid & 31;
  const int lh   = lane & 15;
  const int hi   = lane >> 4;
  const int h    = blockIdx.y;
  const int hk   = h >> 2;
  const int q0   = blockIdx.x * 128;
  const int qw   = q0 + wave * 16;

  const float scale = 0.08838834764831845f;  // 1/sqrt(128)

  // Q fragments (A-layout over d: 4 fragments of K=32)
  v16bf qf[4];
  {
    const __bf16* qp = Q + (size_t)(qw + lh) * HID + h * HDIM;
#pragma unroll
    for (int f = 0; f < 4; ++f) {
      const int kb = f * 32 + hi * 16;
#pragma unroll
      for (int j = 0; j < 8; ++j) { qf[f][j] = qp[kb + j]; qf[f][8 + j] = qp[kb + 8 + j]; }
    }
  }

  v8f oacc[8];
#pragma unroll
  for (int j = 0; j < 8; ++j)
#pragma unroll
    for (int r = 0; r < 8; ++r) oacc[j][r] = 0.f;
  float mrow[8], lrow[8];
#pragma unroll
  for (int r = 0; r < 8; ++r) { mrow[r] = -3.0e38f; lrow[r] = 0.f; }

  const int kend = q0 + 128;  // causal frontier
  for (int kb = 0; kb < kend; kb += 32) {
    // K tile via TDM: 32 rows x 128 elems (256B = 64 DW; pad 4 DW -> iv=5, amt=3)
    if (wave == 0)
      tdm_load_tile_bf16(lds_off_of(lK),
                         Kv + (size_t)kb * (N_KV * HDIM) + hk * HDIM,
                         N_KV * HDIM, S_LEN, HDIM, 32,
                         (unsigned long long)(N_KV * HDIM), 5, 3);
    // V tile transposed by threads (pure bf16 copy)
    {
      const int r = tid >> 3;
      const int c = (tid & 7) * 16;
      const __bf16* vp = Vv + (size_t)(kb + r) * (N_KV * HDIM) + hk * HDIM + c;
#pragma unroll
      for (int j = 0; j < 16; ++j) lVt[(c + j) * VP + r] = vp[j];
    }
    if (wave == 0) __builtin_amdgcn_s_wait_tensorcnt(0);
    __syncthreads();

    // scores S = Q K^T (16x32 per wave)
    v8f sacc[2];
#pragma unroll
    for (int t = 0; t < 2; ++t) {
#pragma unroll
      for (int r = 0; r < 8; ++r) sacc[t][r] = 0.f;
#pragma unroll
      for (int f = 0; f < 4; ++f) {
        v16bf kf;
        const __bf16* p = &lK[(t * 16 + lh) * KP + f * 32 + hi * 16];
#pragma unroll
        for (int j = 0; j < 16; ++j) kf[j] = p[j];
        sacc[t] = __builtin_amdgcn_wmma_f32_16x16x32_bf16(
            false, qf[f], false, kf, (short)0, sacc[t], false, false);
      }
    }

    // scale + causal mask + online softmax (row = qw + hi*8 + r)
#pragma unroll
    for (int r = 0; r < 8; ++r) {
      const int qrow = qw + hi * 8 + r;
#pragma unroll
      for (int t = 0; t < 2; ++t) {
        const int key = kb + t * 16 + lh;
        sacc[t][r] = (key > qrow) ? -3.0e38f : sacc[t][r] * scale;
      }
      float mx = fmaxf(sacc[0][r], sacc[1][r]);
#pragma unroll
      for (int m = 1; m < 16; m <<= 1) mx = fmaxf(mx, __shfl_xor(mx, m, 32));
      const float mnew = fmaxf(mrow[r], mx);
      float rs = 0.f;
#pragma unroll
      for (int t = 0; t < 2; ++t) {
        const float pe = __expf(sacc[t][r] - mnew);
        sacc[t][r] = pe;
        rs += pe;
      }
#pragma unroll
      for (int m = 1; m < 16; m <<= 1) rs += __shfl_xor(rs, m, 32);
      const float corr = __expf(mrow[r] - mnew);
      lrow[r] = lrow[r] * corr + rs;
      mrow[r] = mnew;
#pragma unroll
      for (int j = 0; j < 8; ++j) oacc[j][r] *= corr;
    }

    // P: C-layout -> A-layout via per-wave LDS round trip
#pragma unroll
    for (int r = 0; r < 8; ++r)
#pragma unroll
      for (int t = 0; t < 2; ++t)
        lP[wave][(hi * 8 + r) * VP + t * 16 + lh] = (__bf16)sacc[t][r];
    __syncthreads();   // uniform trip count across waves

    v16bf pf;
    {
      const __bf16* p = &lP[wave][lh * VP + hi * 16];
#pragma unroll
      for (int j = 0; j < 8; ++j) { pf[j] = p[j]; pf[8 + j] = p[8 + j]; }
    }

    // O += P V (8 d-tiles, contraction over 32 keys)
#pragma unroll
    for (int j = 0; j < 8; ++j) {
      v16bf vf;
      const __bf16* p = &lVt[(j * 16 + lh) * VP + hi * 16];
#pragma unroll
      for (int jj = 0; jj < 16; ++jj) vf[jj] = p[jj];
      oacc[j] = __builtin_amdgcn_wmma_f32_16x16x32_bf16(
          false, pf, false, vf, (short)0, oacc[j], false, false);
    }
    __syncthreads();   // protect lK/lVt before next staging
  }

  // normalize, store bf16 [S, NH*HD]
#pragma unroll
  for (int j = 0; j < 8; ++j)
#pragma unroll
    for (int r = 0; r < 8; ++r) {
      const int row = qw + hi * 8 + r;
      const int col = h * HDIM + j * 16 + lh;
      O[(size_t)row * HID + col] = (__bf16)(oacc[j][r] / lrow[r]);
    }
}

// ===========================================================================
// Launch: convert -> QKV gemms -> RoPE -> flash -> output gemm.
// ===========================================================================
extern "C" void kernel_launch(void* const* d_in, const int* in_sizes, int n_in,
                              void* d_out, int out_size, void* d_ws, size_t ws_size,
                              hipStream_t stream) {
  const float* hidden = (const float*)d_in[0];
  const float* Wq     = (const float*)d_in[1];
  const float* Wk     = (const float*)d_in[2];
  const float* Wv     = (const float*)d_in[3];
  const float* Wo     = (const float*)d_in[4];
  // d_in[5] attention_mask: unused (causal applied analytically)
  const int*   pos    = (const int*)d_in[6];
  float*       out    = (float*)d_out;

  __bf16* hbf = (__bf16*)d_ws;
  __bf16* wqT = hbf + (size_t)S_LEN * HID;          // [2048,2048]
  __bf16* wkT = wqT + (size_t)HID * HID;            // [512,2048]
  __bf16* wvT = wkT + (size_t)(N_KV * HDIM) * HID;
  __bf16* woT = wvT + (size_t)(N_KV * HDIM) * HID;  // [2048,2048]
  __bf16* qbf = woT + (size_t)HID * HID;
  __bf16* kbf = qbf + (size_t)S_LEN * HID;
  __bf16* vbf = kbf + (size_t)S_LEN * N_KV * HDIM;
  __bf16* aob = vbf + (size_t)S_LEN * N_KV * HDIM;

  const dim3 blk(256);
  const int NKVD = N_KV * HDIM;

  cvt_bf16<<<(S_LEN * HID / 4 + 255) / 256, blk, 0, stream>>>(hidden, hbf, S_LEN * HID);
  transpose_cvt_bf16<<<dim3(HID / 32, HID / 32),  blk, 0, stream>>>(Wq, wqT, HID, HID);
  transpose_cvt_bf16<<<dim3(NKVD / 32, HID / 32), blk, 0, stream>>>(Wk, wkT, HID, NKVD);
  transpose_cvt_bf16<<<dim3(NKVD / 32, HID / 32), blk, 0, stream>>>(Wv, wvT, HID, NKVD);
  transpose_cvt_bf16<<<dim3(HID / 32, HID / 32),  blk, 0, stream>>>(Wo, woT, HID, HID);

  gemm_bf16_tdm<__bf16><<<dim3(HID / TN,  S_LEN / TM), blk, 0, stream>>>(
      hbf, wqT, qbf, S_LEN, HID, HID);
  gemm_bf16_tdm<__bf16><<<dim3(NKVD / TN, S_LEN / TM), blk, 0, stream>>>(
      hbf, wkT, kbf, S_LEN, NKVD, HID);
  gemm_bf16_tdm<__bf16><<<dim3(NKVD / TN, S_LEN / TM), blk, 0, stream>>>(
      hbf, wvT, vbf, S_LEN, NKVD, HID);

  rope_kernel<<<(S_LEN * N_HEADS * 64 + 255) / 256, 256, 0, stream>>>(qbf, pos, N_HEADS);
  rope_kernel<<<(S_LEN * N_KV   * 64 + 255) / 256, 256, 0, stream>>>(kbf, pos, N_KV);

  flash_attn_wmma<<<dim3(S_LEN / 128, N_HEADS), blk, 0, stream>>>(qbf, kbf, vbf, aob);

  gemm_bf16_tdm<float><<<dim3(HID / TN, S_LEN / TM), blk, 0, stream>>>(
      aob, woT, out, S_LEN, HID, HID);
}
